// CrossAttentionBlock_2362232013219
// MI455X (gfx1250) — matrix operand order repaired and checked by reading, not compiled, not run on hardware
//
#include <hip/hip_runtime.h>
#include <hip/hip_bf16.h>

#define B_  4
#define C_  256
#define S_  2304
#define NH  8
#define HD  32

typedef __attribute__((ext_vector_type(16))) _Float16 v16h;
typedef __attribute__((ext_vector_type(8)))  _Float16 v8h;
typedef __attribute__((ext_vector_type(8)))  float    v8f;

static __device__ inline v16h cat16(v8h lo, v8h hi) {
  return __builtin_shufflevector(lo, hi, 0,1,2,3,4,5,6,7,8,9,10,11,12,13,14,15);
}

// A fragment: 16(M) x 32(K) f16 tile, row-major, leading dim ld (in halves).
// Lane l<16: row l, K 0-7 & 16-23 ; lane l>=16: row l-16, K 8-15 & 24-31.
static __device__ inline v16h load_a_frag(const _Float16* base, int ld, int lane) {
  int lr = lane & 15, hf = lane >> 4;
  const _Float16* r = base + (size_t)lr * ld;
  v8h lo = *(const v8h*)(r + hf * 8);
  v8h hi = *(const v8h*)(r + 16 + hf * 8);
  return cat16(lo, hi);
}
// B fragment from Bt (N x K row-major): lane l<16: col l, K 0-7 & 16-23; lane>=16: col l-16, K 8-15 & 24-31.
static __device__ inline v16h load_b_frag(const _Float16* base, int ld, int lane) {
  int lr = lane & 15, hf = lane >> 4;
  const _Float16* r = base + (size_t)lr * ld + hf * 8;
  v8h lo = *(const v8h*)(r);
  v8h hi = *(const v8h*)(r + 16);
  return cat16(lo, hi);
}

static __device__ inline v8f wmma_f16(v16h a, v16h b, v8f c) {
  return __builtin_amdgcn_wmma_f32_16x16x32_f16(false, a, false, b, (short)0, c, false, false);
}

// ---------------- elementwise f32 -> f16 ----------------
__global__ void k_cvt(const float* __restrict__ in, _Float16* __restrict__ out, int n) {
  int i = blockIdx.x * blockDim.x + threadIdx.x;
  if (i < n) out[i] = (_Float16)in[i];
}

// ---------------- [B,C,S] f32  ->  [B,S,C] f16 (token-major) ----------------
__global__ void k_transpose_cvt(const float* __restrict__ in, _Float16* __restrict__ out) {
  __shared__ float tile[32][33];
  int s0 = blockIdx.x * 32, c0 = blockIdx.y * 32, b = blockIdx.z;
  int tx = threadIdx.x, ty = threadIdx.y;
  #pragma unroll
  for (int i = 0; i < 4; ++i) {
    int cc = ty + 8 * i;
    tile[cc][tx] = in[((size_t)(b * C_ + c0 + cc)) * S_ + s0 + tx];
  }
  __syncthreads();
  #pragma unroll
  for (int i = 0; i < 4; ++i) {
    int rr = ty + 8 * i;
    out[((size_t)(b * S_ + s0 + rr)) * C_ + c0 + tx] = (_Float16)tile[tx][rr];
  }
}

// ---------------- generic WMMA GEMM: D[M,N](f16) = A[M,256] * Bt[N,256]^T + bias ----
// block: 256 thr (8 waves, 4x2), tile M=128 x N=64, K panel 32.
// bias_mode 0: bias[n] ; 1: bias[m]
__global__ void k_gemm_nt(const _Float16* __restrict__ A, int lda, size_t sAb,
                          const _Float16* __restrict__ Bt, int ldb, size_t sBb,
                          _Float16* __restrict__ D, int ldd, size_t sDb,
                          const float* __restrict__ bias, int bias_mode) {
  __shared__ _Float16 sA[128 * 32];
  __shared__ _Float16 sB[64 * 32];
  int t = threadIdx.x, lane = t & 31, w = t >> 5;
  int wm = w & 3, wn = w >> 2;
  int mbase = blockIdx.y * 128, nbase = blockIdx.x * 64, batch = blockIdx.z;
  const _Float16* Ab = A + batch * sAb;
  const _Float16* Bb = Bt + batch * sBb;
  _Float16* Db = D + batch * sDb;

  v8f acc[2][2];
  #pragma unroll
  for (int i = 0; i < 2; ++i)
    #pragma unroll
    for (int j = 0; j < 2; ++j)
      acc[i][j] = (v8f){0.f,0.f,0.f,0.f,0.f,0.f,0.f,0.f};

  for (int k0 = 0; k0 < 256; k0 += 32) {
    __syncthreads();
    #pragma unroll
    for (int i = 0; i < 2; ++i) {          // A: 128 rows x 4 chunks of 16B
      int c = t + i * 256;
      int row = c >> 2, q = c & 3;
      *(uint4*)&sA[row * 32 + q * 8] =
          *(const uint4*)&Ab[(size_t)(mbase + row) * lda + k0 + q * 8];
    }
    {                                      // B: 64 rows x 4 chunks
      int row = t >> 2, q = t & 3;
      *(uint4*)&sB[row * 32 + q * 8] =
          *(const uint4*)&Bb[(size_t)(nbase + row) * ldb + k0 + q * 8];
    }
    __syncthreads();
    v16h af[2], bf[2];
    #pragma unroll
    for (int mt = 0; mt < 2; ++mt) af[mt] = load_a_frag(&sA[(wm * 32 + mt * 16) * 32], 32, lane);
    #pragma unroll
    for (int nt = 0; nt < 2; ++nt) bf[nt] = load_b_frag(&sB[(wn * 32 + nt * 16) * 32], 32, lane);
    #pragma unroll
    for (int mt = 0; mt < 2; ++mt)
      #pragma unroll
      for (int nt = 0; nt < 2; ++nt)
        acc[mt][nt] = wmma_f16(af[mt], bf[nt], acc[mt][nt]);
  }

  int lr = lane & 15, hf = lane >> 4;
  #pragma unroll
  for (int mt = 0; mt < 2; ++mt)
    #pragma unroll
    for (int nt = 0; nt < 2; ++nt) {
      int n = nbase + wn * 32 + nt * 16 + lr;
      float bn = (bias_mode == 0) ? bias[n] : 0.f;
      #pragma unroll
      for (int r = 0; r < 8; ++r) {
        int m = mbase + wm * 32 + mt * 16 + hf * 8 + r;
        float v = acc[mt][nt][r] + ((bias_mode == 0) ? bn : bias[m]);
        Db[(size_t)m * ldd + n] = (_Float16)v;
      }
    }
}

// ---------------- flash attention ----------------
// Q,K: [B,S,256] f16 token-major (head h = cols h*32..). Vt: [B,256,S] f16.
// grid (18, 32): x = 128-row query block, y = b*8+h. 8 waves, 16 q-rows each.
__global__ void k_attn(const _Float16* __restrict__ Qg, const _Float16* __restrict__ Kg,
                       const _Float16* __restrict__ Vtg, _Float16* __restrict__ Og) {
  __shared__ _Float16 sK[32 * 32];     // [key][d]
  __shared__ _Float16 sV[32 * 32];     // [d][key]
  __shared__ _Float16 sP[8][16 * 32];  // per-wave P tile [q][key]
  const float scale = 0.17677669529663687f; // 1/sqrt(32)

  int t = threadIdx.x, lane = t & 31, w = t >> 5;
  int lr = lane & 15, hf = lane >> 4;
  int bh = blockIdx.y, b = bh >> 3, h = bh & 7;
  int qbase = blockIdx.x * 128 + w * 16;

  // Q A-frag, loaded once straight from global (rows contiguous in K=d).
  v16h aq = load_a_frag(Qg + ((size_t)(b * S_ + qbase)) * C_ + h * HD, C_, lane);

  float mrow[8], lsum[8];
  #pragma unroll
  for (int r = 0; r < 8; ++r) { mrow[r] = -3.0e38f; lsum[r] = 0.f; }
  v8f o0 = (v8f){0.f,0.f,0.f,0.f,0.f,0.f,0.f,0.f};
  v8f o1 = o0;
  const v8f vz = o0;

  for (int kt = 0; kt < S_ / 32; ++kt) {
    __syncthreads();
    if (t < 128) {                       // K tile: 32 rows x 64B
      int row = t >> 2, q = t & 3;
      *(uint4*)&sK[row * 32 + q * 8] =
          *(const uint4*)&Kg[((size_t)(b * S_ + kt * 32 + row)) * C_ + h * HD + q * 8];
    } else {                             // V^T tile: 32 d-rows x 64B
      int t2 = t - 128, row = t2 >> 2, q = t2 & 3;
      *(uint4*)&sV[row * 32 + q * 8] =
          *(const uint4*)&Vtg[((size_t)(b * C_ + h * HD + row)) * S_ + kt * 32 + q * 8];
    }
    __syncthreads();

    v16h bk0 = load_b_frag(&sK[0], 32, lane);
    v16h bk1 = load_b_frag(&sK[16 * 32], 32, lane);
    v8f s0 = wmma_f16(aq, bk0, vz);
    v8f s1 = wmma_f16(aq, bk1, vz);

    float p0[8], p1[8];
    #pragma unroll
    for (int r = 0; r < 8; ++r) {
      float v0 = s0[r] * scale, v1 = s1[r] * scale;
      float tm = fmaxf(v0, v1);
      #pragma unroll
      for (int mk = 8; mk >= 1; mk >>= 1) tm = fmaxf(tm, __shfl_xor(tm, mk, 32));
      float mnew = fmaxf(mrow[r], tm);
      float alpha = __expf(mrow[r] - mnew);
      p0[r] = __expf(v0 - mnew);
      p1[r] = __expf(v1 - mnew);
      float rs = p0[r] + p1[r];
      #pragma unroll
      for (int mk = 8; mk >= 1; mk >>= 1) rs += __shfl_xor(rs, mk, 32);
      lsum[r] = lsum[r] * alpha + rs;
      mrow[r] = mnew;
      o0[r] = o0[r] * alpha;
      o1[r] = o1[r] * alpha;
    }
    // P: register(C-frag layout) -> LDS -> A-frag
    #pragma unroll
    for (int r = 0; r < 8; ++r) {
      sP[w][(hf * 8 + r) * 32 + lr]      = (_Float16)p0[r];
      sP[w][(hf * 8 + r) * 32 + 16 + lr] = (_Float16)p1[r];
    }
    __syncthreads();
    v16h ap  = load_a_frag(&sP[w][0], 32, lane);
    v16h bv0 = load_b_frag(&sV[0], 32, lane);
    v16h bv1 = load_b_frag(&sV[16 * 32], 32, lane);
    o0 = wmma_f16(ap, bv0, o0);
    o1 = wmma_f16(ap, bv1, o1);
  }

  #pragma unroll
  for (int r = 0; r < 8; ++r) {
    float inv = 1.f / lsum[r];
    int s = qbase + hf * 8 + r;
    size_t base = ((size_t)(b * S_ + s)) * C_ + h * HD;
    Og[base + lr]      = (_Float16)(o0[r] * inv);
    Og[base + 16 + lr] = (_Float16)(o1[r] * inv);
  }
}

// ---------------- O-projection + residual + LayerNorm + transposed store ----------
// block tile M=32 x N=256 (full channel row). grid (72, B).
__global__ void k_oproj_ln(const _Float16* __restrict__ Ag, const _Float16* __restrict__ Wo16,
                           const float* __restrict__ bo, const float* __restrict__ xres,
                           const float* __restrict__ lnw, const float* __restrict__ lnb,
                           float* __restrict__ out) {
  __shared__ _Float16 sA[32 * 32];
  __shared__ _Float16 sB[256 * 32];
  __shared__ float yt[32][257];
  __shared__ float redS[32][8], redQ[32][8];
  __shared__ float smean[32], sinv[32];

  int t = threadIdx.x, lane = t & 31, w = t >> 5;
  int lr = lane & 15, hf = lane >> 4;
  int mbase = blockIdx.x * 32, b = blockIdx.y;

  v8f acc[2][2];
  #pragma unroll
  for (int i = 0; i < 2; ++i)
    #pragma unroll
    for (int j = 0; j < 2; ++j)
      acc[i][j] = (v8f){0.f,0.f,0.f,0.f,0.f,0.f,0.f,0.f};

  for (int k0 = 0; k0 < 256; k0 += 32) {
    __syncthreads();
    if (t < 128) {                       // A: 32 rows x 4 chunks
      int row = t >> 2, q = t & 3;
      *(uint4*)&sA[row * 32 + q * 8] =
          *(const uint4*)&Ag[((size_t)(b * S_ + mbase + row)) * C_ + k0 + q * 8];
    }
    #pragma unroll
    for (int i = 0; i < 4; ++i) {        // B: 256 rows x 4 chunks
      int c = t + i * 256;
      int row = c >> 2, q = c & 3;
      *(uint4*)&sB[row * 32 + q * 8] =
          *(const uint4*)&Wo16[(size_t)row * 256 + k0 + q * 8];
    }
    __syncthreads();
    v16h af[2], bf[2];
    #pragma unroll
    for (int mt = 0; mt < 2; ++mt) af[mt] = load_a_frag(&sA[(mt * 16) * 32], 32, lane);
    #pragma unroll
    for (int nt = 0; nt < 2; ++nt) bf[nt] = load_b_frag(&sB[(w * 32 + nt * 16) * 32], 32, lane);
    #pragma unroll
    for (int mt = 0; mt < 2; ++mt)
      #pragma unroll
      for (int nt = 0; nt < 2; ++nt)
        acc[mt][nt] = wmma_f16(af[mt], bf[nt], acc[mt][nt]);
  }

  // epilogue: + bias + residual, stage into yt
  #pragma unroll
  for (int mt = 0; mt < 2; ++mt)
    #pragma unroll
    for (int nt = 0; nt < 2; ++nt) {
      int c = w * 32 + nt * 16 + lr;
      float bn = bo[c];
      int srow0 = mt * 16 + hf * 8;
      const float* xp = xres + ((size_t)(b * C_ + c)) * S_ + mbase + srow0;
      float4 x0 = *(const float4*)xp;
      float4 x1 = *(const float4*)(xp + 4);
      float xv[8] = {x0.x, x0.y, x0.z, x0.w, x1.x, x1.y, x1.z, x1.w};
      #pragma unroll
      for (int r = 0; r < 8; ++r)
        yt[srow0 + r][c] = acc[mt][nt][r] + bn + xv[r];
    }
  __syncthreads();

  {                                      // per-row mean/var over 256 channels
    int row = t & 31, chunk = t >> 5;
    float s1 = 0.f, s2 = 0.f;
    #pragma unroll
    for (int j = 0; j < 32; ++j) {
      float v = yt[row][chunk * 32 + j];
      s1 += v; s2 += v * v;
    }
    redS[row][chunk] = s1; redQ[row][chunk] = s2;
  }
  __syncthreads();
  if (t < 32) {
    float s1 = 0.f, s2 = 0.f;
    #pragma unroll
    for (int j = 0; j < 8; ++j) { s1 += redS[t][j]; s2 += redQ[t][j]; }
    float mu = s1 * (1.f / 256.f);
    float var = s2 * (1.f / 256.f) - mu * mu;
    smean[t] = mu;
    sinv[t] = rsqrtf(var + 1e-5f);
  }
  __syncthreads();
  {                                      // normalize + transposed store (coalesced over s)
    int row = t & 31, chunk = t >> 5;
    float mu = smean[row], iv = sinv[row];
    int s = mbase + row;
    for (int j = 0; j < 32; ++j) {
      int c = chunk * 32 + j;
      float v = (yt[row][c] - mu) * iv * lnw[c] + lnb[c];
      out[((size_t)(b * C_ + c)) * S_ + s] = v;
    }
  }
}

extern "C" void kernel_launch(void* const* d_in, const int* in_sizes, int n_in,
                              void* d_out, int out_size, void* d_ws, size_t ws_size,
                              hipStream_t stream) {
  const float* x   = (const float*)d_in[0];
  const float* ctx = (const float*)d_in[1];
  const float* Wq  = (const float*)d_in[2];
  const float* bq  = (const float*)d_in[3];
  const float* Wk  = (const float*)d_in[4];
  const float* bk  = (const float*)d_in[5];
  const float* Wv  = (const float*)d_in[6];
  const float* bv  = (const float*)d_in[7];
  const float* Wo  = (const float*)d_in[8];
  const float* bo  = (const float*)d_in[9];
  const float* lnw = (const float*)d_in[10];
  const float* lnb = (const float*)d_in[11];
  float* out = (float*)d_out;

  const size_t E = (size_t)B_ * S_ * C_;          // 2,359,296 elements
  const size_t need = (6 * E + 4 * 65536) * sizeof(_Float16);
  if (ws_size < need) return;

  _Float16* Xt   = (_Float16*)d_ws;
  _Float16* Ct   = Xt + E;
  _Float16* Qg   = Ct + E;
  _Float16* Kg   = Qg + E;
  _Float16* Vt   = Kg + E;
  _Float16* At   = Vt + E;
  _Float16* Wq16 = At + E;
  _Float16* Wk16 = Wq16 + 65536;
  _Float16* Wv16 = Wk16 + 65536;
  _Float16* Wo16 = Wv16 + 65536;

  dim3 tb(32, 8);
  dim3 tg(S_ / 32, C_ / 32, B_);
  k_transpose_cvt<<<tg, tb, 0, stream>>>(x, Xt);
  k_transpose_cvt<<<tg, tb, 0, stream>>>(ctx, Ct);
  k_cvt<<<256, 256, 0, stream>>>(Wq, Wq16, 65536);
  k_cvt<<<256, 256, 0, stream>>>(Wk, Wk16, 65536);
  k_cvt<<<256, 256, 0, stream>>>(Wv, Wv16, 65536);
  k_cvt<<<256, 256, 0, stream>>>(Wo, Wo16, 65536);

  const size_t sTok = (size_t)S_ * C_;
  // Q = Xt @ Wq^T + bq  -> [B,S,C] f16
  k_gemm_nt<<<dim3(C_ / 64, S_ / 128, B_), 256, 0, stream>>>(
      Xt, C_, sTok, Wq16, C_, 0, Qg, C_, sTok, bq, 0);
  // K = Ct @ Wk^T + bk  -> [B,S,C] f16
  k_gemm_nt<<<dim3(C_ / 64, S_ / 128, B_), 256, 0, stream>>>(
      Ct, C_, sTok, Wk16, C_, 0, Kg, C_, sTok, bk, 0);
  // V^T = Wv @ Ct^T + bv(rows) -> [B,C,S] f16
  k_gemm_nt<<<dim3(S_ / 64, C_ / 128, B_), 256, 0, stream>>>(
      Wv16, C_, 0, Ct, C_, sTok, Vt, S_, sTok, bv, 1);

  k_attn<<<dim3(S_ / 128, B_ * NH), 256, 0, stream>>>(Qg, Kg, Vt, At);

  k_oproj_ln<<<dim3(S_ / 32, B_), 256, 0, stream>>>(At, Wo16, bo, x, lnw, lnb, out);
}
